// LowRankStructureLearner_51307679318378
// MI455X (gfx1250) — matrix operand order repaired and checked by reading, not compiled, not run on hardware
//
#include <hip/hip_runtime.h>

// ---------------------------------------------------------------------------
// LowRankStructureLearner for MI455X (gfx1250, wave32, WMMA + TDM).
// All matmuls use v_wmma_f32_16x16x32_f16 (f16 operands, f32 accum).
// attn [B,H,N,N] f32 is streamed to d_out exactly once (output-BW bound).
// V head-slice is staged into LDS by the Tensor Data Mover, overlapped with
// the score/top-k phases. Intermediates kept as f16 in workspace (~15 MB).
// ---------------------------------------------------------------------------

typedef __attribute__((ext_vector_type(16))) _Float16 v16h;
typedef __attribute__((ext_vector_type(8)))  float    v8f;
typedef __attribute__((ext_vector_type(4)))  unsigned u32x4;
typedef __attribute__((ext_vector_type(8)))  unsigned u32x8;

#define D_    256
#define RANK_ 64
#define H_    8
#define DH_   32
#define TOPK_ 32
#define B_    2
#define N_    2048
#define BN_   (B_ * N_)

static __device__ __forceinline__ v8f wmma16(v16h a, v16h b, v8f c) {
  // (neg_a, A, neg_b, B, c_mod, C, reuse_a, reuse_b)
  return __builtin_amdgcn_wmma_f32_16x16x32_f16(false, a, false, b, (short)0, c,
                                                false, false);
}

// ---- fragment loaders (contiguous-k order, same for A and B, wave32) -------------------
// 16-bit A (16x32 MxK): lane = hi*16+row; VGPR g, elem e -> K = g*2 + e +
// hi*16
static __device__ __forceinline__ v16h afrag_f16(const _Float16* A, int ld,
                                                 int m0, int k0) {
  int lane = threadIdx.x & 31;
  int row = lane & 15, hi = lane >> 4;
  const _Float16* p = A + (long)(m0 + row) * ld + k0 + hi * 16;
  v16h a;
#pragma unroll
  for (int g = 0; g < 8; ++g) {
    int kk = g * 2;
    a[2 * g + 0] = p[kk + 0];
    a[2 * g + 1] = p[kk + 1];
  }
  return a;
}

static __device__ __forceinline__ v16h afrag_f32(const float* A, int ld,
                                                 int m0, int k0) {
  int lane = threadIdx.x & 31;
  int row = lane & 15, hi = lane >> 4;
  const float* p = A + (long)(m0 + row) * ld + k0 + hi * 16;
  v16h a;
#pragma unroll
  for (int g = 0; g < 8; ++g) {
    int kk = g * 2;
    a[2 * g + 0] = (_Float16)p[kk + 0];
    a[2 * g + 1] = (_Float16)p[kk + 1];
  }
  return a;
}

// 16-bit B (32x16 KxN): lanes 0-15 hold K=0..15 (col=lane), lanes 16-31 hold
// K=16..31; element j -> K = hi*16 + j.
// Row-major f32 weight W[K][Ncols], tile at (k0,n0):
static __device__ __forceinline__ v16h bfrag_wf32(const float* W, int ldn,
                                                  int k0, int n0) {
  int lane = threadIdx.x & 31;
  int col = lane & 15, hi = lane >> 4;
  const float* p = W + (long)(k0 + hi * 16) * ldn + n0 + col;
  v16h b;
#pragma unroll
  for (int j = 0; j < 16; ++j) b[j] = (_Float16)p[(long)j * ldn];
  return b;
}

// Row-major f16 source S[K][ld]:
static __device__ __forceinline__ v16h bfrag_rm_f16(const _Float16* S, int ld,
                                                    int k0, int n0) {
  int lane = threadIdx.x & 31;
  int col = lane & 15, hi = lane >> 4;
  const _Float16* p = S + (long)(k0 + hi * 16) * ld + n0 + col;
  v16h b;
#pragma unroll
  for (int j = 0; j < 16; ++j) b[j] = p[(long)j * ld];
  return b;
}

// Transposed source: B[k][n] = S[n0+n][k0+k] (contiguous 16 halves per lane,
// e.g. K^T for Q@K^T and el^T for el@el^T):
static __device__ __forceinline__ v16h bfrag_tr_f16(const _Float16* S, int ld,
                                                    int n0, int k0) {
  int lane = threadIdx.x & 31;
  int col = lane & 15, hi = lane >> 4;
  const _Float16* p = S + (long)(n0 + col) * ld + k0 + hi * 16;
  v16h b;
#pragma unroll
  for (int j = 0; j < 16; ++j) b[j] = p[j];
  return b;
}

// C/D 16x16 f32: lane col=lane&15, rows M = r + hi*8
static __device__ __forceinline__ void store_tile_f16(_Float16* C, int ld,
                                                      int m0, int n0, v8f acc) {
  int lane = threadIdx.x & 31;
  int col = lane & 15, hi = lane >> 4;
#pragma unroll
  for (int r = 0; r < 8; ++r)
    C[(long)(m0 + r + hi * 8) * ld + n0 + col] = (_Float16)acc[r];
}

// order-preserving float <-> uint mapping (for exact top-k radix search)
static __device__ __forceinline__ unsigned f2o(float f) {
  unsigned u = __float_as_uint(f);
  return (u & 0x80000000u) ? ~u : (u | 0x80000000u);
}
static __device__ __forceinline__ float o2f(unsigned u) {
  return (u & 0x80000000u) ? __uint_as_float(u & 0x7FFFFFFFu)
                           : __uint_as_float(~u);
}

// ---------------------------------------------------------------------------
// Generic low-rank GEMM stage: C(f16)[M,Ncols] = A[M,K] @ W(f32)[K,Ncols]
// One wave per 16x16 output tile, 8 waves per block.
// ---------------------------------------------------------------------------
template <bool A_F32>
__global__ __launch_bounds__(256) void lowrank_gemm(const void* Ain,
                                                    const float* W,
                                                    _Float16* C, int M, int K,
                                                    int Ncols) {
  int wave = threadIdx.x >> 5;
  int tiles_n = Ncols >> 4;
  int tile = blockIdx.x * (blockDim.x >> 5) + wave;
  int m0 = (tile / tiles_n) << 4;
  int n0 = (tile % tiles_n) << 4;
  if (m0 >= M) return;
  v8f acc = {};
  for (int k0 = 0; k0 < K; k0 += 32) {
    v16h a = A_F32 ? afrag_f32((const float*)Ain, K, m0, k0)
                   : afrag_f16((const _Float16*)Ain, K, m0, k0);
    v16h b = bfrag_wf32(W, Ncols, k0, n0);
    acc = wmma16(a, b, acc);
  }
  store_tile_f16(C, Ncols, m0, n0, acc);
}

// ---------------------------------------------------------------------------
// Fused attention: TDM-stage V -> scores -> exact top-k -> softmax -> attn out
// -> ctx = attn @ v(LDS).  One block per (b, h, 16-query tile); 8 waves.
// Dynamic LDS: 16x2048 f32 score/prob panel (128 KB) + 8 KB reduce scratch
//            + 2048x32 f16 V stage (128 KB) = 264 KB (fits 320 KB WGP LDS).
// ---------------------------------------------------------------------------
__global__ __launch_bounds__(256) void attn_topk_kernel(
    const _Float16* q, const _Float16* k, const _Float16* v, float* attn_out,
    _Float16* ctx_out) {
  extern __shared__ float smem[];
  float* sc = smem;                                   // [16][N_]
  float* red = smem + 16 * N_;                        // [8][256]
  _Float16* vlds = (_Float16*)(smem + 16 * N_ + 8 * 256);  // [N_][DH_]

  int bh = blockIdx.x / (N_ / 16);  // b*H + h
  int qt = blockIdx.x % (N_ / 16);
  int b = bh / H_, hh = bh % H_;
  int n0 = qt * 16;
  int wave = threadIdx.x >> 5;
  int lane = threadIdx.x & 31;

  const _Float16* qbase = q + (long)b * N_ * D_ + hh * DH_;
  const _Float16* kbase = k + (long)b * N_ * D_ + hh * DH_;
  const _Float16* vbase = v + (long)b * N_ * D_ + hh * DH_;

  // ---- TDM: async-stage V head-slice [N_,DH_] (row stride D_) into LDS ----
  // D# per CDNA5 ISA 8.3/8.4: 2D tile, data_size=2B, tile 32 x 2048.
  if (wave == 0) {
    unsigned long long ga = (unsigned long long)(uintptr_t)vbase;
    unsigned ldsa = (unsigned)(uintptr_t)vlds;  // generic low 32b = LDS offset
    u32x4 g0;
    g0[0] = 1u;                                          // count=1, user mode
    g0[1] = ldsa;                                        // lds_addr
    g0[2] = (unsigned)ga;                                // global_addr[31:0]
    g0[3] = (unsigned)((ga >> 32) & 0x1FFFFFFu) | (2u << 30);  // [56:32]|type=2
    u32x8 g1;
    g1[0] = (1u << 16);          // workgroup_mask=0, data_size=1 (2 bytes)
    g1[1] = (DH_ & 0xFFFFu) << 16;   // tensor_dim0 = 32   (bits 79:48)
    g1[2] = (N_ & 0xFFFFu) << 16;    // tensor_dim1 = 2048 (bits 111:80)
    g1[3] = ((unsigned)DH_ << 16);   // tile_dim0 = 32     (bits 127:112)
    g1[4] = (unsigned)N_;            // tile_dim1 = 2048   (bits 143:128)
    g1[5] = (unsigned)D_;            // tensor_dim0_stride = 256 (bits 207:160)
    g1[6] = 0u;
    g1[7] = 0u;
    asm volatile("tensor_load_to_lds %0, %1" ::"s"(g0), "s"(g1) : "memory");
  }

  // ---- phase 1: score panel, one WMMA (K = DH = 32) per 16-key tile -------
  v16h aq = afrag_f16(qbase, D_, n0, 0);
  const float scale = 0.17677669529663687f;  // 1/sqrt(32)
  for (int j = wave; j < N_ / 16; j += 8) {
    if (j + 8 < N_ / 16)
      __builtin_prefetch(kbase + (long)(j + 8) * 16 * D_, 0, 1);
    v16h bk = bfrag_tr_f16(kbase, D_, j * 16, 0);
    v8f acc = {};
    acc = wmma16(aq, bk, acc);
    int col = lane & 15, hi = lane >> 4;
#pragma unroll
    for (int r = 0; r < 8; ++r)
      sc[(r + hi * 8) * N_ + j * 16 + col] = acc[r] * scale;
  }
  __syncthreads();

  // ---- phase 2: per-row exact kth-largest + softmax, 2 rows per wave ------
  // Each lane owns 64 values, held in registers as order-preserving uints.
  for (int rr = 0; rr < 2; ++rr) {
    int row = wave * 2 + rr;
    float* srow = sc + row * N_;
    unsigned u[64];
#pragma unroll
    for (int t = 0; t < 64; ++t) u[t] = f2o(srow[lane + 32 * t]);
    // row max (always a survivor, so softmax shift is the plain max)
    unsigned um = 0u;
#pragma unroll
    for (int t = 0; t < 64; ++t) um = (u[t] > um) ? u[t] : um;
#pragma unroll
    for (int off = 16; off > 0; off >>= 1) {
      unsigned o = (unsigned)__shfl_xor((int)um, off, 32);
      um = (o > um) ? o : um;
    }
    float m = o2f(um);
    // radix binary search: max thr with count(u >= thr) >= TOPK (exact kth)
    unsigned thr = 0u;
    for (int bit = 31; bit >= 0; --bit) {
      unsigned cand = thr | (1u << bit);
      int c = 0;
#pragma unroll
      for (int t = 0; t < 64; ++t) c += (u[t] >= cand) ? 1 : 0;
#pragma unroll
      for (int off = 16; off > 0; off >>= 1) c += __shfl_xor(c, off, 32);
      if (c >= TOPK_) thr = cand;
    }
    // softmax over survivors (non-survivors are exactly 0 in f32)
    float s = 0.f;
#pragma unroll
    for (int t = 0; t < 64; ++t)
      if (u[t] >= thr) s += __expf(o2f(u[t]) - m);
#pragma unroll
    for (int off = 16; off > 0; off >>= 1) s += __shfl_xor(s, off, 32);
    float inv = 1.f / s;
    float* arow = attn_out + (((long)bh * N_) + n0 + row) * N_;
#pragma unroll
    for (int t = 0; t < 64; ++t) {
      float p = (u[t] >= thr) ? __expf(o2f(u[t]) - m) * inv : 0.f;
      srow[lane + 32 * t] = p;  // probs stay in LDS for phase 3
      arow[lane + 32 * t] = p;  // stream attn to d_out
    }
  }
  __builtin_amdgcn_s_wait_tensorcnt(0);  // V stage complete
  __syncthreads();

  // ---- phase 3: ctx[16,32] = probs[16,2048] @ vlds[2048,32] ---------------
  // wave -> (col half nh = wave&1, k-slice = wave>>1 of 4)
  int nh = wave & 1;
  int slice = wave >> 1;
  v8f acc = {};
  for (int jc = slice; jc < N_ / 32; jc += 4) {
    v16h ap = afrag_f32(sc, N_, 0, jc * 32);              // probs (LDS)
    v16h bv = bfrag_rm_f16(vlds, DH_, jc * 32, nh * 16);  // V (LDS, TDM-staged)
    acc = wmma16(ap, bv, acc);
  }
#pragma unroll
  for (int r = 0; r < 8; ++r) red[wave * 256 + r * 32 + lane] = acc[r];
  __syncthreads();
  if (wave < 2) {  // wave == nh: reduce 4 k-slice partials, emit f16 ctx
    int col = lane & 15, hi = lane >> 4;
#pragma unroll
    for (int r = 0; r < 8; ++r) {
      float s4 = red[(wave + 0) * 256 + r * 32 + lane] +
                 red[(wave + 2) * 256 + r * 32 + lane] +
                 red[(wave + 4) * 256 + r * 32 + lane] +
                 red[(wave + 6) * 256 + r * 32 + lane];
      ctx_out[((long)b * N_ + n0 + r + hi * 8) * D_ + hh * DH_ + wave * 16 +
              col] = (_Float16)s4;
    }
  }
}

// ---------------------------------------------------------------------------
// adjacency = sigmoid(el @ el^T), zero diagonal. One wave per 16x16 tile.
// ---------------------------------------------------------------------------
__global__ __launch_bounds__(256) void adjacency_kernel(const _Float16* el,
                                                        float* adj) {
  int wave = threadIdx.x >> 5, lane = threadIdx.x & 31;
  int tile = blockIdx.x * (blockDim.x >> 5) + wave;
  const int tn = N_ / 16;  // 128
  int bb = tile / (tn * tn);
  int t2 = tile % (tn * tn);
  int m0 = (t2 / tn) * 16, n0 = (t2 % tn) * 16;
  const _Float16* e = el + (long)bb * N_ * D_;
  v8f acc = {};
#pragma unroll
  for (int k0 = 0; k0 < D_; k0 += 32) {
    v16h a = afrag_f16(e, D_, m0, k0);
    v16h bt = bfrag_tr_f16(e, D_, n0, k0);
    acc = wmma16(a, bt, acc);
  }
  int col = lane & 15, hi = lane >> 4;
  float* arow = adj + (long)bb * N_ * N_;
#pragma unroll
  for (int r = 0; r < 8; ++r) {
    int m = m0 + r + hi * 8, n = n0 + col;
    float sg = 1.f / (1.f + __expf(-acc[r]));
    arow[(long)m * N_ + n] = (m == n) ? 0.f : sg;
  }
}

// ---------------------------------------------------------------------------
extern "C" void kernel_launch(void* const* d_in, const int* in_sizes, int n_in,
                              void* d_out, int out_size, void* d_ws,
                              size_t ws_size, hipStream_t stream) {
  (void)in_sizes; (void)n_in; (void)out_size; (void)ws_size;
  const float* x    = (const float*)d_in[0];
  const float* ne_u = (const float*)d_in[1];
  const float* ne_v = (const float*)d_in[2];
  const float* wq_u = (const float*)d_in[3];
  const float* wq_v = (const float*)d_in[4];
  const float* wk_u = (const float*)d_in[5];
  const float* wk_v = (const float*)d_in[6];
  const float* wv_u = (const float*)d_in[7];
  const float* wv_v = (const float*)d_in[8];
  const float* wo_u = (const float*)d_in[9];
  const float* wo_v = (const float*)d_in[10];
  const float* es_u = (const float*)d_in[11];
  const float* es_v = (const float*)d_in[12];

  float* adj  = (float*)d_out;                       // [B,N,N]
  float* attn = (float*)d_out + (long)B_ * N_ * N_;  // [B,H,N,N]

  // workspace carve-up (f16 intermediates, 256B-aligned, ~15 MB total)
  char* ws = (char*)d_ws;
  size_t off = 0;
  auto alloc_h = [&](size_t elems) {
    _Float16* p = (_Float16*)(ws + off);
    off += ((elems * sizeof(_Float16) + 255) & ~(size_t)255);
    return p;
  };
  _Float16* tbuf = alloc_h((size_t)BN_ * RANK_);  // low-rank inner T
  _Float16* hbuf = alloc_h((size_t)BN_ * D_);     // node embedding h
  _Float16* qb   = alloc_h((size_t)BN_ * D_);
  _Float16* kb   = alloc_h((size_t)BN_ * D_);
  _Float16* vb   = alloc_h((size_t)BN_ * D_);
  _Float16* ob   = alloc_h((size_t)BN_ * D_);     // attn output ctx
  _Float16* h2b  = alloc_h((size_t)BN_ * D_);
  _Float16* elb  = alloc_h((size_t)BN_ * D_);

  auto lowrank2 = [&](const void* A, bool a_f32, const float* U,
                      const float* V, _Float16* outp) {
    const int tiles1 = (BN_ / 16) * (RANK_ / 16);  // 1024
    if (a_f32)
      lowrank_gemm<true><<<tiles1 / 8, 256, 0, stream>>>(A, U, tbuf, BN_, D_,
                                                         RANK_);
    else
      lowrank_gemm<false><<<tiles1 / 8, 256, 0, stream>>>(A, U, tbuf, BN_, D_,
                                                          RANK_);
    const int tiles2 = (BN_ / 16) * (D_ / 16);     // 4096
    lowrank_gemm<false><<<tiles2 / 8, 256, 0, stream>>>(tbuf, V, outp, BN_,
                                                        RANK_, D_);
  };

  // h = lowrank(x); q/k/v = lowrank(h)
  lowrank2(x, true, ne_u, ne_v, hbuf);
  lowrank2(hbuf, false, wq_u, wq_v, qb);
  lowrank2(hbuf, false, wk_u, wk_v, kb);
  lowrank2(hbuf, false, wv_u, wv_v, vb);

  // fused scores/top-k/softmax/ctx
  // dynamic LDS: 128KB score panel + 8KB scratch + 128KB TDM V stage
  const size_t shmem =
      (size_t)(16 * N_ + 8 * 256) * sizeof(float) + (size_t)N_ * DH_ * 2;
  hipFuncSetAttribute(reinterpret_cast<const void*>(attn_topk_kernel),
                      hipFuncAttributeMaxDynamicSharedMemorySize, (int)shmem);
  attn_topk_kernel<<<B_ * H_ * (N_ / 16), 256, shmem, stream>>>(qb, kb, vb,
                                                                attn, ob);

  // h2 = lowrank(ctx, wo); el = lowrank(h2, es)
  lowrank2(ob, false, wo_u, wo_v, h2b);
  lowrank2(h2b, false, es_u, es_v, elb);

  // adjacency = sigmoid(el @ el^T) with zero diagonal
  adjacency_kernel<<<(B_ * (N_ / 16) * (N_ / 16)) / 8, 256, 0, stream>>>(elb,
                                                                         adj);
}
